// MJAVEInteraction_79061757985279
// MI455X (gfx1250) — compile-verified, matrix operand order checked by reading, not run on hardware
//
#include <hip/hip_runtime.h>
#include <hip/hip_bf16.h>
#include <math.h>

// ---------------- problem dims (fixed by the reference) ----------------
constexpr int Bsz  = 512;
constexpr int Lq   = 46;    // seq len
constexpr int Rr   = 49;    // image regions
constexpr int TXTD = 768;
constexpr int IMGD = 2048;
constexpr int HDD  = 200;   // all_head_size
constexpr int NL   = 28;    // labels
constexpr int NB   = 55;    // BIO tags

typedef __attribute__((ext_vector_type(16))) __bf16          v16bf;
typedef __attribute__((ext_vector_type(8)))  float           v8f;
typedef __attribute__((ext_vector_type(8)))  unsigned short  u16x8;
typedef __attribute__((ext_vector_type(4)))  unsigned short  u16x4;
typedef __attribute__((ext_vector_type(4)))  float           f32x4;

// fp32 -> bf16 (round-to-nearest-even), raw u16 for LDS staging
__device__ __forceinline__ unsigned short f2bf_u16(float f) {
  unsigned u = __builtin_bit_cast(unsigned, f);
  unsigned r = u + 0x7FFFu + ((u >> 16) & 1u);
  return (unsigned short)(r >> 16);
}

struct BfPair { u16x8 lo, hi; };

// Per-lane fragment = two contiguous 16B runs (ISA 7.12.2 K-interleave):
// halves [kbase, kbase+8) and [kbase+16, kbase+24). Row stride is 80B
// (multiple of 16) so both loads are ds_load_b128-able.
__device__ __forceinline__ v16bf frag_from_lds(const unsigned short* row, int kbase) {
  BfPair p;
  p.lo = *(const u16x8*)(row + kbase);
  p.hi = *(const u16x8*)(row + kbase + 16);
  return __builtin_bit_cast(v16bf, p);
}

// =====================================================================
// Generic GEMM:  C[M,N] = A[M,K] * W[K,N] (+ bias[N])
// bf16 WMMA 16x16x32, fp32 accumulate. Tile 128x64, BK=32, 256 threads.
// 8 waves in 4(M) x 2(N); each wave owns a 32x32 patch = 4 accumulators,
// fed by 2 A-fragments + 2 B-fragments per K step (4 WMMAs / step).
// =====================================================================
#define GM_BM 128
#define GM_BN 64
#define GM_BK 32
#define GM_LDA 40   // halves; 80B row stride (16B-aligned rows, conflict padding)
#define GM_LDB 40

__global__ __launch_bounds__(256) void gemm_bf16_wmma(
    const float* __restrict__ A, const float* __restrict__ W,
    const float* __restrict__ bias, float* __restrict__ C,
    int M, int N, int K)
{
  __shared__ alignas(16) unsigned short sA[GM_BM * GM_LDA];  // 10240 B
  __shared__ alignas(16) unsigned short sB[GM_BN * GM_LDB];  // 5120 B, transposed [n][k]

  const int tid   = threadIdx.x;
  const int lane  = tid & 31;
  const int wave  = tid >> 5;             // 0..7
  const int mTile = blockIdx.x * GM_BM;
  const int nTile = blockIdx.y * GM_BN;
  const int wm = (wave & 3) * 32;         // wave row offset in tile
  const int wn = (wave >> 2) * 32;        // wave col offset in tile

  const int rc = lane & 15;               // row (A) / col (B) within 16
  const int kb = (lane >> 4) * 8;         // K interleave per ISA 7.12.2

  v8f acc00 = {0.f,0.f,0.f,0.f,0.f,0.f,0.f,0.f};
  v8f acc01 = acc00, acc10 = acc00, acc11 = acc00;

  for (int k0 = 0; k0 < K; k0 += GM_BK) {
    // ---- stage A tile (128x32) : coalesced float4 loads, packed u16x4 stores
    // K is always a multiple of 4 here, so a float4 is in/out of range as a unit.
    for (int idx = tid; idx < (GM_BM * GM_BK) / 4; idx += 256) {
      int r  = idx >> 3;                 // 0..127
      int c4 = (idx & 7) << 2;           // 0,4,..,28
      int gm = mTile + r, gk = k0 + c4;
      int sm = gm < M ? gm : 0;
      int sk = gk < K ? gk : 0;
      f32x4 v = *(const f32x4*)(A + (size_t)sm * (size_t)K + sk);
      bool ok = (gm < M) && (gk < K);
      u16x4 h;
#pragma unroll
      for (int j = 0; j < 4; ++j) h[j] = f2bf_u16(ok ? v[j] : 0.f);
      *(u16x4*)(sA + r * GM_LDA + c4) = h;
    }
    // ---- stage B tile (32x64) transposed into sB[n][k]
    for (int idx = tid; idx < GM_BK * GM_BN; idx += 256) {
      int r = idx >> 6, c = idx & 63;    // r = k row, c = n col (coalesced reads)
      int gk = k0 + r, gn = nTile + c;
      int sk = gk < K ? gk : 0;
      int sn = gn < N ? gn : 0;
      float v = W[(size_t)sk * (size_t)N + sn];
      v = (gk < K && gn < N) ? v : 0.f;
      sB[c * GM_LDB + r] = f2bf_u16(v);
    }
    __syncthreads();

    v16bf a0 = frag_from_lds(sA + (wm + rc)      * GM_LDA, kb);
    v16bf a1 = frag_from_lds(sA + (wm + 16 + rc) * GM_LDA, kb);
    v16bf b0 = frag_from_lds(sB + (wn + rc)      * GM_LDB, kb);
    v16bf b1 = frag_from_lds(sB + (wn + 16 + rc) * GM_LDB, kb);

    acc00 = __builtin_amdgcn_wmma_f32_16x16x32_bf16(false, a0, false, b0, (short)0, acc00, false, false);
    acc01 = __builtin_amdgcn_wmma_f32_16x16x32_bf16(false, a0, false, b1, (short)0, acc01, false, false);
    acc10 = __builtin_amdgcn_wmma_f32_16x16x32_bf16(false, a1, false, b0, (short)0, acc10, false, false);
    acc11 = __builtin_amdgcn_wmma_f32_16x16x32_bf16(false, a1, false, b1, (short)0, acc11, false, false);
    __syncthreads();
  }

  // C/D layout: VGPR i -> row i (lanes 0-15) / i+8 (lanes 16-31), col = lane&15
  const int rsub = ((lane >> 4) << 3);
  {
    const v8f* accs[4] = { &acc00, &acc01, &acc10, &acc11 };
#pragma unroll
    for (int mi = 0; mi < 2; ++mi)
#pragma unroll
      for (int ni = 0; ni < 2; ++ni) {
        const v8f& a = *accs[mi * 2 + ni];
        int col = nTile + wn + ni * 16 + rc;
        if (col < N) {
          float bv = bias ? bias[col] : 0.f;
          int row0 = mTile + wm + mi * 16 + rsub;
#pragma unroll
          for (int i = 0; i < 8; ++i) {
            int gr = row0 + i;
            if (gr < M) C[(size_t)gr * (size_t)N + col] = a[i] + bv;
          }
        }
      }
  }
}

// =====================================================================
// txt self-attention, one block per batch element
// =====================================================================
__global__ __launch_bounds__(256) void txt_attn_kernel(
    const float* __restrict__ Q, const float* __restrict__ Ks,
    const float* __restrict__ V, const float* __restrict__ mask,
    float* __restrict__ Out)
{
  int b = blockIdx.x;
  __shared__ float sc[Lq * Lq];
  const float scale = 0.070710678118654752f;   // 1/sqrt(200)
  const float* q = Q  + (size_t)b * Lq * HDD;
  const float* k = Ks + (size_t)b * Lq * HDD;
  const float* v = V  + (size_t)b * Lq * HDD;

  for (int p = threadIdx.x; p < Lq * Lq; p += 256) {
    int qi = p / Lq, ki = p % Lq;
    const float* qr = q + (size_t)qi * HDD;
    const float* kr = k + (size_t)ki * HDD;
    float s = 0.f;
    for (int d = 0; d < HDD; ++d) s += qr[d] * kr[d];
    sc[p] = s * scale + (1.f - mask[(size_t)b * Lq + ki]) * (-10000.f);
  }
  __syncthreads();
  for (int qi = threadIdx.x; qi < Lq; qi += 256) {
    float m = -1e30f;
    for (int j = 0; j < Lq; ++j) m = fmaxf(m, sc[qi * Lq + j]);
    float sum = 0.f;
    for (int j = 0; j < Lq; ++j) { float e = expf(sc[qi * Lq + j] - m); sc[qi * Lq + j] = e; sum += e; }
    float inv = 1.f / sum;
    for (int j = 0; j < Lq; ++j) sc[qi * Lq + j] *= inv;
  }
  __syncthreads();
  for (int p = threadIdx.x; p < Lq * HDD; p += 256) {
    int qi = p / HDD, d = p % HDD;
    float s = 0.f;
    for (int j = 0; j < Lq; ++j) s += sc[qi * Lq + j] * v[(size_t)j * HDD + d];
    Out[(size_t)b * Lq * HDD + p] = s;
  }
}

// =====================================================================
// cross-modal attention (text queries, image-region kv), keeps probs
// =====================================================================
__global__ __launch_bounds__(256) void img_attn_kernel(
    const float* __restrict__ Q, const float* __restrict__ Ks,
    const float* __restrict__ V, float* __restrict__ probs,
    float* __restrict__ Out)
{
  int b = blockIdx.x;
  __shared__ float sc[Lq * Rr];
  const float scale = 0.070710678118654752f;
  const float* q = Q  + (size_t)b * Lq * HDD;
  const float* k = Ks + (size_t)b * Rr * HDD;
  const float* v = V  + (size_t)b * Rr * HDD;

  for (int p = threadIdx.x; p < Lq * Rr; p += 256) {
    int qi = p / Rr, ri = p % Rr;
    const float* qr = q + (size_t)qi * HDD;
    const float* kr = k + (size_t)ri * HDD;
    float s = 0.f;
    for (int d = 0; d < HDD; ++d) s += qr[d] * kr[d];
    sc[p] = s * scale;
  }
  __syncthreads();
  for (int qi = threadIdx.x; qi < Lq; qi += 256) {
    float m = -1e30f;
    for (int j = 0; j < Rr; ++j) m = fmaxf(m, sc[qi * Rr + j]);
    float sum = 0.f;
    for (int j = 0; j < Rr; ++j) { float e = expf(sc[qi * Rr + j] - m); sc[qi * Rr + j] = e; sum += e; }
    float inv = 1.f / sum;
    for (int j = 0; j < Rr; ++j) sc[qi * Rr + j] *= inv;
  }
  __syncthreads();
  for (int p = threadIdx.x; p < Lq * Rr; p += 256)
    probs[(size_t)b * Lq * Rr + p] = sc[p];
  for (int p = threadIdx.x; p < Lq * HDD; p += 256) {
    int qi = p / HDD, d = p % HDD;
    float s = 0.f;
    for (int j = 0; j < Rr; ++j) s += sc[qi * Rr + j] * v[(size_t)j * HDD + d];
    Out[(size_t)b * Lq * HDD + p] = s;
  }
}

// =====================================================================
// global gate + mm_attn = txt_attn + gg * img_attn ; one block per (b,l)
// =====================================================================
__global__ __launch_bounds__(128) void gg_mm_kernel(
    const float* __restrict__ seq_txt, const float* __restrict__ global_img,
    const float* __restrict__ gg_txt, const float* __restrict__ gg_img,
    const float* __restrict__ txt_attn, const float* __restrict__ img_attn,
    float* __restrict__ mm)
{
  int b = blockIdx.x / Lq, l = blockIdx.x % Lq;
  __shared__ float red[128];
  float part = 0.f;
  const float* st = seq_txt + ((size_t)b * Lq + l) * TXTD;
  for (int i = threadIdx.x; i < TXTD; i += 128) part += st[i] * gg_txt[i];
  const float* gi = global_img + (size_t)b * IMGD;
  for (int i = threadIdx.x; i < IMGD; i += 128) part += gi[i] * gg_img[(size_t)i * Lq + l];
  red[threadIdx.x] = part; __syncthreads();
  for (int s = 64; s > 0; s >>= 1) {
    if (threadIdx.x < s) red[threadIdx.x] += red[threadIdx.x + s];
    __syncthreads();
  }
  float g = 1.f / (1.f + expf(-red[0]));
  size_t off = ((size_t)b * Lq + l) * HDD;
  for (int d = threadIdx.x; d < HDD; d += 128)
    mm[off + d] = txt_attn[off + d] + g * img_attn[off + d];
}

__global__ void mean_kernel(const float* __restrict__ mm, float* __restrict__ mean) {
  int b = blockIdx.x;
  for (int d = threadIdx.x; d < HDD; d += blockDim.x) {
    float s = 0.f;
    for (int l = 0; l < Lq; ++l) s += mm[((size_t)b * Lq + l) * HDD + d];
    mean[(size_t)b * HDD + d] = s * (1.0f / Lq);
  }
}

// label logits + sigmoid; one block per (b, n)
__global__ __launch_bounds__(128) void label_kernel(
    const float* __restrict__ cls, const float* __restrict__ meanmm,
    const float* __restrict__ Wl, const float* __restrict__ bl,
    float* __restrict__ out_logits, float* __restrict__ out_pred,
    float* __restrict__ ws_pred)
{
  int b = blockIdx.x / NL, n = blockIdx.x % NL;
  __shared__ float red[128];
  float s = 0.f;
  for (int i = threadIdx.x; i < TXTD; i += 128)
    s += cls[(size_t)b * TXTD + i] * Wl[(size_t)i * NL + n];
  for (int i = threadIdx.x; i < HDD; i += 128)
    s += meanmm[(size_t)b * HDD + i] * Wl[(size_t)(TXTD + i) * NL + n];
  red[threadIdx.x] = s; __syncthreads();
  for (int st = 64; st > 0; st >>= 1) {
    if (threadIdx.x < st) red[threadIdx.x] += red[threadIdx.x + st];
    __syncthreads();
  }
  if (threadIdx.x == 0) {
    float lg = red[0] + bl[n];
    out_logits[(size_t)b * NL + n] = lg;
    float p = 1.f / (1.f + expf(-lg));
    out_pred[(size_t)b * NL + n] = p;
    ws_pred[(size_t)b * NL + n]  = p;
  }
}

// label injection + tvec = rg_img @ pred  (re-associated regional gate)
__global__ __launch_bounds__(256) void inj_tvec_kernel(
    const float* __restrict__ pred, const float* __restrict__ Winj,
    const float* __restrict__ Wrg, float* __restrict__ inj, float* __restrict__ tvec)
{
  int b = blockIdx.x;
  __shared__ float pl[NL];
  if (threadIdx.x < NL) pl[threadIdx.x] = pred[(size_t)b * NL + threadIdx.x];
  __syncthreads();
  for (int d = threadIdx.x; d < HDD; d += 256) {
    float s = 0.f;
    for (int n = 0; n < NL; ++n) s += pl[n] * Winj[(size_t)n * HDD + d];
    inj[(size_t)b * HDD + d] = s;
  }
  for (int i = threadIdx.x; i < IMGD; i += 256) {
    const float* wr = Wrg + (size_t)i * NL;
    float s = 0.f;
    for (int n = 0; n < NL; ++n) s += wr[n] * pl[n];
    tvec[(size_t)b * IMGD + i] = s;
  }
}

// rg[b,r] = sigmoid(regional[b,r,:] . tvec[b,:]) ; one block per (b,r)
__global__ __launch_bounds__(256) void rg_kernel(
    const float* __restrict__ regional, const float* __restrict__ tvec,
    float* __restrict__ rg)
{
  int b = blockIdx.x / Rr, r = blockIdx.x % Rr;
  __shared__ float red[256];
  const float* reg = regional + ((size_t)b * Rr + r) * IMGD;
  const float* t   = tvec + (size_t)b * IMGD;
  float s = 0.f;
  for (int i = threadIdx.x; i < IMGD; i += 256) s += reg[i] * t[i];
  red[threadIdx.x] = s; __syncthreads();
  for (int st = 128; st > 0; st >>= 1) {
    if (threadIdx.x < st) red[threadIdx.x] += red[threadIdx.x + st];
    __syncthreads();
  }
  if (threadIdx.x == 0) rg[(size_t)b * Rr + r] = 1.f / (1.f + expf(-red[0]));
}

// lrg[b,l,d] = sum_r rg[b,r]*probs[b,l,r]*Vcm[b,r,d] ; one block per (b,l)
__global__ __launch_bounds__(256) void lrg_kernel(
    const float* __restrict__ rg, const float* __restrict__ probs,
    const float* __restrict__ Vcm, float* __restrict__ lrg)
{
  int b = blockIdx.x / Lq, l = blockIdx.x % Lq;
  __shared__ float w[Rr];
  if (threadIdx.x < Rr)
    w[threadIdx.x] = rg[(size_t)b * Rr + threadIdx.x] *
                     probs[((size_t)b * Lq + l) * Rr + threadIdx.x];
  __syncthreads();
  size_t off = ((size_t)b * Lq + l) * HDD;
  for (int d = threadIdx.x; d < HDD; d += 256) {
    float s = 0.f;
    for (int r = 0; r < Rr; ++r) s += w[r] * Vcm[((size_t)b * Rr + r) * HDD + d];
    lrg[off + d] = s;
  }
}

// pre = txtproj + mmproj + lrg + inj(broadcast over L)
__global__ void combine_kernel(
    const float* __restrict__ tp, const float* __restrict__ mp,
    const float* __restrict__ lr, const float* __restrict__ inj,
    float* __restrict__ pre, size_t total)
{
  size_t i = (size_t)blockIdx.x * blockDim.x + threadIdx.x;
  if (i < total) {
    int d = (int)(i % HDD);
    size_t b = i / ((size_t)Lq * HDD);
    pre[i] = tp[i] + mp[i] + lr[i] + inj[b * HDD + d];
  }
}

// row softmax over NB=55; one block per (b,l)
__global__ __launch_bounds__(64) void seq_softmax_kernel(
    const float* __restrict__ logits, float* __restrict__ pred)
{
  int row = blockIdx.x;
  const float* x = logits + (size_t)row * NB;
  __shared__ float sm[64];
  int tid = threadIdx.x;
  float m = -1e30f;
  for (int i = tid; i < NB; i += 64) m = fmaxf(m, x[i]);
  sm[tid] = m; __syncthreads();
  for (int s = 32; s > 0; s >>= 1) { if (tid < s) sm[tid] = fmaxf(sm[tid], sm[tid + s]); __syncthreads(); }
  m = sm[0]; __syncthreads();
  float sum = 0.f;
  for (int i = tid; i < NB; i += 64) sum += expf(x[i] - m);
  sm[tid] = sum; __syncthreads();
  for (int s = 32; s > 0; s >>= 1) { if (tid < s) sm[tid] += sm[tid + s]; __syncthreads(); }
  float inv = 1.f / sm[0];
  for (int i = tid; i < NB; i += 64) pred[(size_t)row * NB + i] = expf(x[i] - m) * inv;
}

__global__ void tail_copy(const float* __restrict__ mask, const int* __restrict__ seqlen,
                          float* __restrict__ om, float* __restrict__ ol,
                          int nMask, int nLen)
{
  int i = blockIdx.x * blockDim.x + threadIdx.x;
  if (i < nMask) om[i] = mask[i];
  if (i < nLen)  ol[i] = (float)seqlen[i];
}

// =====================================================================
// host launcher
// =====================================================================
extern "C" void kernel_launch(void* const* d_in, const int* in_sizes, int n_in,
                              void* d_out, int out_size, void* d_ws, size_t ws_size,
                              hipStream_t stream)
{
  (void)in_sizes; (void)n_in; (void)out_size; (void)ws_size;

  // inputs (setup_inputs dict order; params in insertion order, (W,b) flattened)
  const float* global_img = (const float*)d_in[0];     // [B,IMG]
  const float* regional   = (const float*)d_in[1];     // [B,R,IMG]
  const float* cls        = (const float*)d_in[2];     // [B,TXT]
  const float* seq_txt    = (const float*)d_in[3];     // [B,L,TXT]
  const float* mask       = (const float*)d_in[4];     // [B,L]
  const int*   seqlen     = (const int*)d_in[5];       // [B]
  const float* Wtq = (const float*)d_in[6],  *btq = (const float*)d_in[7];
  const float* Wtk = (const float*)d_in[8],  *btk = (const float*)d_in[9];
  const float* Wtv = (const float*)d_in[10], *btv = (const float*)d_in[11];
  const float* Wcq = (const float*)d_in[12], *bcq = (const float*)d_in[13];
  const float* Wck = (const float*)d_in[14], *bck = (const float*)d_in[15];
  const float* Wcv = (const float*)d_in[16], *bcv = (const float*)d_in[17];
  const float* Wggt = (const float*)d_in[18];          // [TXT,1]
  const float* Wggi = (const float*)d_in[19];          // [IMG,L]
  const float* Wrg  = (const float*)d_in[20];          // [IMG,NL]
  const float* Wl   = (const float*)d_in[21], *bl = (const float*)d_in[22]; // [TXT+HD,NL]
  const float* Wstp = (const float*)d_in[23];          // [TXT,HD]
  const float* Wmmp = (const float*)d_in[24];          // [HD,HD]
  const float* Winj = (const float*)d_in[25];          // [NL,HD]
  const float* Wsp  = (const float*)d_in[26], *bsp = (const float*)d_in[27]; // [HD,NB]

  // output layout (flat, return order)
  float* out       = (float*)d_out;
  float* o_logl    = out;
  float* o_predl   = o_logl   + (size_t)Bsz * NL;
  float* o_logseq  = o_predl  + (size_t)Bsz * NL;
  float* o_predseq = o_logseq + (size_t)Bsz * Lq * NB;
  float* o_mask    = o_predseq+ (size_t)Bsz * Lq * NB;
  float* o_len     = o_mask   + (size_t)Bsz * Lq;

  // workspace (~163 MB of fp32), with lifetime-based aliasing
  const size_t nBLH = (size_t)Bsz * Lq * HDD;   // 4,710,400
  const size_t nBRH = (size_t)Bsz * Rr * HDD;   // 5,017,600
  float* ws   = (float*)d_ws;
  float* buf0 = ws;                 // Qtxt   -> mm_attn
  float* buf1 = buf0 + nBLH;        // Ktxt   -> TxtProj
  float* buf2 = buf1 + nBLH;        // Vtxt   -> MMproj
  float* buf3 = buf2 + nBLH;        // Qcm    -> lrg
  float* buf4 = buf3 + nBLH;        // txt_attn -> pre_seq
  float* buf5 = buf4 + nBLH;        // img_attn
  float* Kcm  = buf5 + nBLH;        // [B,R,HD]
  float* Vcm  = Kcm + nBRH;         // [B,R,HD]
  float* probs= Vcm + nBRH;         // [B,L,R]
  float* meanm= probs + (size_t)Bsz * Lq * Rr;  // [B,HD]
  float* predl= meanm + (size_t)Bsz * HDD;      // [B,NL]
  float* inj  = predl + (size_t)Bsz * NL;       // [B,HD]
  float* tvec = inj  + (size_t)Bsz * HDD;       // [B,IMG]
  float* rgb  = tvec + (size_t)Bsz * IMGD;      // [B,R]

  const int MBL = Bsz * Lq;   // 23552 (184 * 128)
  const int MBR = Bsz * Rr;   // 25088 (196 * 128)

  auto gemm = [&](const float* A, const float* W, const float* bias, float* C,
                  int M, int N, int K) {
    dim3 g((M + GM_BM - 1) / GM_BM, (N + GM_BN - 1) / GM_BN);
    gemm_bf16_wmma<<<g, 256, 0, stream>>>(A, W, bias, C, M, N, K);
  };

  // 1) projections (WMMA GEMMs)
  gemm(seq_txt,  Wtq, btq, buf0, MBL, HDD, TXTD);   // Qtxt
  gemm(seq_txt,  Wtk, btk, buf1, MBL, HDD, TXTD);   // Ktxt
  gemm(seq_txt,  Wtv, btv, buf2, MBL, HDD, TXTD);   // Vtxt
  gemm(seq_txt,  Wcq, bcq, buf3, MBL, HDD, TXTD);   // Qcm
  gemm(regional, Wck, bck, Kcm,  MBR, HDD, IMGD);   // Kcm
  gemm(regional, Wcv, bcv, Vcm,  MBR, HDD, IMGD);   // Vcm

  // 2) attentions
  txt_attn_kernel<<<Bsz, 256, 0, stream>>>(buf0, buf1, buf2, mask, buf4);
  img_attn_kernel<<<Bsz, 256, 0, stream>>>(buf3, Kcm, Vcm, probs, buf5);

  // 3) global gate + fuse  (mm_attn -> buf0)
  gg_mm_kernel<<<Bsz * Lq, 128, 0, stream>>>(seq_txt, global_img, Wggt, Wggi,
                                             buf4, buf5, buf0);

  // 4) label head
  mean_kernel<<<Bsz, 256, 0, stream>>>(buf0, meanm);
  label_kernel<<<Bsz * NL, 128, 0, stream>>>(cls, meanm, Wl, bl,
                                             o_logl, o_predl, predl);
  inj_tvec_kernel<<<Bsz, 256, 0, stream>>>(predl, Winj, Wrg, inj, tvec);
  rg_kernel<<<Bsz * Rr, 256, 0, stream>>>(regional, tvec, rgb);

  // 5) regional re-weighted context (lrg -> buf3)
  lrg_kernel<<<Bsz * Lq, 256, 0, stream>>>(rgb, probs, Vcm, buf3);

  // 6) sequence head
  gemm(seq_txt, Wstp, nullptr, buf1, MBL, HDD, TXTD);  // seq_txt_proj
  gemm(buf0,    Wmmp, nullptr, buf2, MBL, HDD, HDD);   // mm_attn @ seq_mm_proj
  {
    size_t total = (size_t)MBL * HDD;
    int grid = (int)((total + 255) / 256);
    combine_kernel<<<grid, 256, 0, stream>>>(buf1, buf2, buf3, inj, buf4, total);
  }
  gemm(buf4, Wsp, bsp, o_logseq, MBL, NB, HDD);        // logits_seq
  seq_softmax_kernel<<<Bsz * Lq, 64, 0, stream>>>(o_logseq, o_predseq);

  // 7) passthrough outputs
  {
    int nMask = Bsz * Lq;
    int grid  = (nMask + 255) / 256;
    tail_copy<<<grid, 256, 0, stream>>>(mask, seqlen, o_mask, o_len, nMask, Bsz);
  }
}
